// pSp_65738769432620
// MI455X (gfx1250) — matrix operand order repaired and checked by reading, not compiled, not run on hardware
//
#include <hip/hip_runtime.h>
#include <stdint.h>

// ---------------------------------------------------------------------------
// StyleGAN2 block for MI455X (gfx1250, wave32).
// Compute-bound (~194 GFLOP vs ~0.2GB HBM) -> f16 WMMA (v_wmma_f32_16x16x32_f16)
// implicit-GEMM convs; weights staged via Tensor Data Mover (tensor_load_to_lds
// + s_wait_tensorcnt), activations staged transposed into LDS for contiguous
// B-fragment reads.
// ---------------------------------------------------------------------------

typedef _Float16 half_t;
typedef __attribute__((ext_vector_type(8)))  _Float16 v8h;
typedef __attribute__((ext_vector_type(16))) _Float16 v16h;
typedef __attribute__((ext_vector_type(8)))  float    v8f;
typedef __attribute__((ext_vector_type(4)))  unsigned int v4u;
typedef __attribute__((ext_vector_type(8)))  int      v8i;
typedef __attribute__((ext_vector_type(4)))  int      v4i;

#ifndef __has_builtin
#define __has_builtin(x) 0
#endif

#if defined(__gfx1250__) && __has_builtin(__builtin_amdgcn_tensor_load_to_lds) && \
    __has_builtin(__builtin_amdgcn_s_wait_tensorcnt)
#define USE_TDM 1
#else
#define USE_TDM 0
#endif

// ---------------------------------------------------------------------------
// TDM helpers: 3D tile [tile2 cout][9 taps][32 cin] of f16 weights, layout
// [b][cout][tap][cin] in global (stride0=512 elems between taps, stride1=4608
// between cout rows). tensor_dims set large (no OOB clipping needed).
// ---------------------------------------------------------------------------
#if USE_TDM
__device__ __forceinline__ void tdm_load_w(unsigned ldsByteOff, const half_t* g,
                                           unsigned tile2) {
  unsigned long long ga = (unsigned long long)(uintptr_t)g;
  v4u g0 = { 1u,                                   // count=1, user descriptor
             ldsByteOff,                           // lds_addr
             (unsigned)(ga & 0xFFFFFFFFu),         // global_addr[31:0]
             (unsigned)(((ga >> 32) & 0x01FFFFFFull) | 0x80000000u) }; // type=2
  v8i g1 = { 0x00010000,                           // data_size = 2 bytes
             0,                                    // no atomic barrier
             0x10,                                 // tensor_dim0 = 1<<20 (hi)
             (int)(0x10u | (32u << 16)),           // tensor_dim1 hi | tile_dim0=32
             (int)(9u | (tile2 << 16)),            // tile_dim1=9 | tile_dim2
             512,                                  // tensor_dim0_stride = 512
             (int)(4608u << 16),                   // tensor_dim1_stride = 4608
             0 };
  v4i g2 = { 0x100000, 0, 0, 0 };                  // tensor_dim2 large
  v4i g3 = { 0, 0, 0, 0 };
#if __clang_major__ >= 23
  v8i gz = { 0, 0, 0, 0, 0, 0, 0, 0 };
  __builtin_amdgcn_tensor_load_to_lds(g0, g1, g2, g3, gz, 0);
#else
  __builtin_amdgcn_tensor_load_to_lds(g0, g1, g2, g3, 0);
#endif
}
#endif

// Fallback weight staging (identical LDS layout to the TDM tile walk order).
__device__ __forceinline__ void stage_w_manual(half_t* dst, const half_t* wgbase,
                                               int cinBase, int rows) {
  const int n = rows * 288;  // rows * 9 taps * 32 cin
  for (int e = threadIdx.x; e < n; e += 256) {
    int co = e / 288;
    int rem = e - co * 288;
    int tap = rem >> 5;
    int cc = rem & 31;
    dst[e] = wgbase[((size_t)co * 9 + tap) * 512 + cinBase + cc];
  }
}

// ---------------------------------------------------------------------------
// WMMA fragment helpers (16x16x32 f16 -> f32).
// A (16x32 f16): lanes 0-15 hold row M=lane, K 0..15; lanes 16-31 hold
// K 16..31  -> two 16B LDS reads from [row][tap][cin] layout.
// B (32x16 f16): lanes 0-15 hold N=lane, K 0..15; lanes 16-31 K 16..31 -> one
// contiguous 32B read from transposed [pix][cin] LDS layout.
// ---------------------------------------------------------------------------
__device__ __forceinline__ v16h load_a_frag(const half_t* Wl, int row, int tap,
                                            int halfsel) {
  const half_t* p = Wl + ((row * 9 + tap) * 32 + halfsel * 16);
  v8h lo = *(const v8h*)p;
  v8h hi = *(const v8h*)(p + 8);
  return __builtin_shufflevector(lo, hi, 0, 1, 2, 3, 4, 5, 6, 7,
                                 8, 9, 10, 11, 12, 13, 14, 15);
}

__device__ __forceinline__ v8f wmma16(v16h a, v16h b, v8f c) {
  return __builtin_amdgcn_wmma_f32_16x16x32_f16(false, a, false, b,
                                                (short)0, c, false, false);
}

// ---------------------------------------------------------------------------
// Small utility kernels.
// ---------------------------------------------------------------------------
__global__ __launch_bounds__(256) void zero_kernel(unsigned* __restrict__ p, int n) {
  int i = blockIdx.x * 256 + threadIdx.x;
  if (i < n) p[i] = 0u;
}

__global__ __launch_bounds__(256) void cast_kernel(const float* __restrict__ s,
                                                   half_t* __restrict__ d, int n) {
  int i = blockIdx.x * 256 + threadIdx.x;
  if (i < n) d[i] = (half_t)s[i];
}

// Per-(b,cout) modulation + demodulation -> f16 weights [b][cout][tap][cin].
// flip=1 bakes the conv_transpose kernel flip (reads w[2-ky][2-kx]).
__global__ __launch_bounds__(256) void modw_kernel(const float* __restrict__ w,
                                                   const float* __restrict__ s,
                                                   half_t* __restrict__ wmout,
                                                   int flip) {
  __shared__ float red[256];
  const int cout = blockIdx.x;
  const int b = blockIdx.y;
  const int tid = threadIdx.x;
  const float scale = 0.014731391274719739f;  // 1/sqrt(512*9)
  float sum = 0.f;
  for (int e = tid; e < 4608; e += 256) {
    int tap = e >> 9, cin = e & 511;
    int ky = tap / 3, kx = tap - ky * 3;
    int rky = flip ? 2 - ky : ky, rkx = flip ? 2 - kx : kx;
    float v = scale * w[(((size_t)cout * 512 + cin) * 3 + rky) * 3 + rkx] *
              s[b * 512 + cin];
    sum += v * v;
  }
  red[tid] = sum;
  __syncthreads();
  for (int o = 128; o > 0; o >>= 1) {
    if (tid < o) red[tid] += red[tid + o];
    __syncthreads();
  }
  const float d = rsqrtf(red[0] + 1e-8f);
  for (int e = tid; e < 4608; e += 256) {
    int tap = e >> 9, cin = e & 511;
    int ky = tap / 3, kx = tap - ky * 3;
    int rky = flip ? 2 - ky : ky, rkx = flip ? 2 - kx : kx;
    float v = scale * w[(((size_t)cout * 512 + cin) * 3 + rky) * 3 + rkx] *
              s[b * 512 + cin];
    wmout[((size_t)(b * 512 + cout) * 9 + tap) * 512 + cin] = (half_t)(v * d);
  }
}

// ---------------------------------------------------------------------------
// conv1: transpose conv (stride 2, k=3) as phase-decomposed implicit GEMM.
// Block: 8 waves = 4(M=16) x 2(N=16): 64 cout x 32 output-phase pixels.
// Grid: x = oy*2+px (130), y = cout/64 (8), z = batch (8).
// Output: raw pre-blur y1 [B,512,65,65] f16 (column 64 pre-zeroed).
// ---------------------------------------------------------------------------
__global__ __launch_bounds__(256) void conv1_wmma_kernel(
    const half_t* __restrict__ xh,   // [B,512,32,32]
    const half_t* __restrict__ wm,   // [B,512,9,512] (flipped)
    half_t* __restrict__ y1) {       // [B,512,65,65]
  constexpr int WB = 64 * 9 * 32;    // 18432 elems (36864 B)
  constexpr int ITOFF = WB;
  constexpr int ITN = 2 * 33 * 32;   // 2112 elems
  __shared__ __align__(64) half_t smem[WB + ITN];

  const int tid = threadIdx.x;
  const int lane = tid & 31;
  const int wv = tid >> 5;
  const int halfsel = lane >> 4;
  const int ln = lane & 15;
  const int coutOff = (wv >> 1) * 16;   // 0,16,32,48
  const int nOff = (wv & 1) * 16;
  const int rp = blockIdx.x;
  const int oy = rp >> 1;
  const int px = rp & 1;
  const int py = oy & 1;
  const int coutBase = blockIdx.y * 64;
  const int b = blockIdx.z;
  const half_t* wgbase = wm + (size_t)(b * 512 + coutBase) * 9 * 512;

  for (int e = tid; e < WB + ITN; e += 256) smem[e] = (half_t)0;
  __syncthreads();

  v8f acc = {0.f, 0.f, 0.f, 0.f, 0.f, 0.f, 0.f, 0.f};

  for (int kc = 0; kc < 16; ++kc) {
    const int cinBase = kc * 32;
#if USE_TDM
    if (tid < 32) tdm_load_w(0u, wgbase + cinBase, 64u);
#else
    stage_w_manual(smem, wgbase, cinBase, 64);
#endif
    // Stage input rows (transposed [row][xi][cin]) while TDM runs.
    const half_t* xb = xh + (size_t)(b * 512 + cinBase) * 1024;
    for (int e = tid; e < ITN; e += 256) {
      int cc = e & 31;
      int xi = (e >> 5) % 33;
      int r = e / 1056;
      int iy = (py == 0) ? ((oy >> 1) - 1 + r) : ((r == 0) ? ((oy - 1) >> 1) : -1);
      int ix = xi - 1;
      half_t v = (half_t)0;
      if (iy >= 0 && iy < 32 && ix >= 0 && ix < 32)
        v = xb[(size_t)cc * 1024 + iy * 32 + ix];
      smem[ITOFF + (r * 33 + xi) * 32 + cc] = v;
    }
#if USE_TDM
    if (tid < 32) __builtin_amdgcn_s_wait_tensorcnt(0);
#endif
    __syncthreads();

    const int m = nOff + ln;
    const int nky = (py == 0) ? 2 : 1;
    const int nkx = (px == 0) ? 2 : 1;
    for (int ai = 0; ai < nky; ++ai) {
      int ky = (py == 0) ? (ai * 2) : 1;
      int r = (py == 0) ? ai : 0;
      for (int bi = 0; bi < nkx; ++bi) {
        int kx = (px == 0) ? (bi * 2) : 1;
        int xi = (kx == 0) ? m : (m + 1);
        int tap = ky * 3 + kx;
        v16h af = load_a_frag(smem, coutOff + ln, tap, halfsel);
        const half_t* bp = smem + ITOFF + (r * 33 + xi) * 32 + halfsel * 16;
        v16h bf = *(const v16h*)bp;
        acc = wmma16(af, bf, acc);
      }
    }
    __syncthreads();
  }

#pragma unroll
  for (int i = 0; i < 8; ++i) {
    int cout = coutBase + coutOff + halfsel * 8 + i;
    int m2 = nOff + ln;
    int ox = px + 2 * m2;
    y1[((size_t)(b * 512 + cout) * 65 + oy) * 65 + ox] = (half_t)acc[i];
  }
}

// Blur (4x4 FIR, pad(1,1)) + noise + bias + LeakyReLU*sqrt(2)  ->  y1b f16.
__global__ __launch_bounds__(256) void blur_noise_act_kernel(
    const half_t* __restrict__ y1, const float* __restrict__ nw,
    const float* __restrict__ noise, const float* __restrict__ bias,
    half_t* __restrict__ out) {
  int i = blockIdx.x * 256 + threadIdx.x;
  if (i >= 8 * 512 * 64 * 64) return;
  int x = i & 63, y = (i >> 6) & 63, c = (i >> 12) & 511, b = i >> 21;
  const float kt[4] = {1.f, 3.f, 3.f, 1.f};
  const half_t* src = y1 + (size_t)(b * 512 + c) * 4225;
  float acc = 0.f;
#pragma unroll
  for (int ty = 0; ty < 4; ++ty) {
    int ys = y + ty - 1;
    if (ys < 0 || ys > 64) continue;
#pragma unroll
    for (int tx = 0; tx < 4; ++tx) {
      int xs = x + tx - 1;
      if (xs < 0 || xs > 64) continue;
      acc += (kt[ty] * kt[tx]) * (float)src[ys * 65 + xs];
    }
  }
  acc *= (1.f / 16.f);
  acc += nw[0] * noise[(size_t)b * 4096 + y * 64 + x];
  acc += bias[c];
  acc = (acc >= 0.f ? acc : 0.2f * acc) * 1.41421356237309f;
  out[i] = (half_t)acc;
}

// ---------------------------------------------------------------------------
// conv2: plain 3x3 implicit GEMM. Block: 8 waves = 2(M=16) x 4(N=16):
// 32 cout x full 64-pixel row. Grid: x=row (64), y=cout/32 (16), z=batch (8).
// TDM weight tiles double-buffered (overlap with input staging + compute).
// Fused epilogue: noise + bias + LeakyReLU*sqrt(2) -> y2 f16.
// ---------------------------------------------------------------------------
__global__ __launch_bounds__(256) void conv2_wmma_kernel(
    const half_t* __restrict__ xin,  // y1b [B,512,64,64]
    const half_t* __restrict__ wm,   // [B,512,9,512]
    const float* __restrict__ nw, const float* __restrict__ noise,
    const float* __restrict__ bias, half_t* __restrict__ out) {
  constexpr int WB = 32 * 9 * 32;    // 9216 elems per buffer (18432 B)
  constexpr int ITOFF = 2 * WB;
  constexpr int ITN = 3 * 66 * 32;   // 6336 elems
  __shared__ __align__(64) half_t smem[2 * WB + ITN];

  const int tid = threadIdx.x;
  const int lane = tid & 31;
  const int wv = tid >> 5;
  const int halfsel = lane >> 4;
  const int ln = lane & 15;
  const int coutOff = (wv >> 2) * 16;   // 0,16
  const int xOff = (wv & 3) * 16;       // 0..48
  const int y0 = blockIdx.x;
  const int coutBase = blockIdx.y * 32;
  const int b = blockIdx.z;
  const half_t* wgbase = wm + (size_t)(b * 512 + coutBase) * 9 * 512;

  for (int e = tid; e < 2 * WB + ITN; e += 256) smem[e] = (half_t)0;
  __syncthreads();

  v8f acc = {0.f, 0.f, 0.f, 0.f, 0.f, 0.f, 0.f, 0.f};

#if USE_TDM
  if (tid < 32) tdm_load_w(0u, wgbase, 32u);  // chunk 0 -> buffer 0
#else
  stage_w_manual(smem, wgbase, 0, 32);
#endif

  for (int kc = 0; kc < 16; ++kc) {
    const int cinBase = kc * 32;
    const int cur = kc & 1;

    // Stage transposed input tile [row3][xi66][cin32] (overlaps current TDM).
    const half_t* xb = xin + (size_t)(b * 512 + cinBase) * 4096;
    if (kc + 1 < 16)
      __builtin_prefetch(xin + (size_t)(b * 512 + cinBase + 32) * 4096 + y0 * 64, 0, 1);
    for (int e = tid; e < ITN; e += 256) {
      int cc = e & 31;
      int xi = (e >> 5) % 66;
      int r = e / 2112;
      int yy = y0 + r - 1;
      int xx = xi - 1;
      half_t v = (half_t)0;
      if (yy >= 0 && yy < 64 && xx >= 0 && xx < 64)
        v = xb[(size_t)cc * 4096 + yy * 64 + xx];
      smem[ITOFF + (r * 66 + xi) * 32 + cc] = v;
    }

#if USE_TDM
    if (tid < 32) {
      if (kc + 1 < 16) {
        tdm_load_w((unsigned)((cur ^ 1) * WB * 2), wgbase + (kc + 1) * 32, 32u);
        __builtin_amdgcn_s_wait_tensorcnt(1);   // current chunk done
      } else {
        __builtin_amdgcn_s_wait_tensorcnt(0);
      }
    }
#else
    if (kc + 1 < 16) stage_w_manual(smem + (cur ^ 1) * WB, wgbase, (kc + 1) * 32, 32);
#endif
    __syncthreads();

    const half_t* Wl = smem + cur * WB;
    const half_t* It = smem + ITOFF;
#pragma unroll
    for (int ty = 0; ty < 3; ++ty) {
#pragma unroll
      for (int tx = 0; tx < 3; ++tx) {
        int tap = ty * 3 + tx;
        v16h af = load_a_frag(Wl, coutOff + ln, tap, halfsel);
        const half_t* bp = It + (ty * 66 + xOff + ln + tx) * 32 + halfsel * 16;
        v16h bf = *(const v16h*)bp;
        acc = wmma16(af, bf, acc);
      }
    }
    __syncthreads();
  }

  const float nwv = nw[0];
#pragma unroll
  for (int i = 0; i < 8; ++i) {
    int cout = coutBase + coutOff + halfsel * 8 + i;
    int x = xOff + ln;
    float v = acc[i];
    v += nwv * noise[(size_t)b * 4096 + y0 * 64 + x];
    v += bias[cout];
    v = (v >= 0.f ? v : 0.2f * v) * 1.41421356237309f;
    out[((size_t)(b * 512 + cout) * 64 + y0) * 64 + x] = (half_t)v;
  }
}

// ToRGB (1x1 modulated conv, no demod) + bias + upsampled skip (bw-bound).
__global__ __launch_bounds__(256) void torgb_skip_kernel(
    const half_t* __restrict__ y2, const float* __restrict__ wrgb,
    const float* __restrict__ srgb, const float* __restrict__ brgb,
    const float* __restrict__ skip, float* __restrict__ out) {
  int t = blockIdx.x * 256 + threadIdx.x;
  if (t >= 8 * 64 * 64) return;
  int x = t & 63, y = (t >> 6) & 63, b = t >> 12;
  const float scale = 0.044194173824159216f;  // 1/sqrt(512)
  float a0 = 0.f, a1 = 0.f, a2 = 0.f;
  const half_t* yb = y2 + (size_t)b * 512 * 4096 + y * 64 + x;
  const float* sb = srgb + b * 512;
  for (int cin = 0; cin < 512; ++cin) {
    float v = (float)yb[(size_t)cin * 4096];
    float sv = sb[cin] * scale * v;
    a0 += sv * wrgb[cin];
    a1 += sv * wrgb[512 + cin];
    a2 += sv * wrgb[1024 + cin];
  }
  float res[3] = {a0 + brgb[0], a1 + brgb[1], a2 + brgb[2]};
  const float kt[4] = {1.f, 3.f, 3.f, 1.f};
#pragma unroll
  for (int c = 0; c < 3; ++c) {
    float su = 0.f;
#pragma unroll
    for (int ty = 0; ty < 4; ++ty) {
      int dy = y + ty - 2;
      if (dy < 0 || (dy & 1)) continue;
      int iy = dy >> 1;
      if (iy >= 32) continue;
#pragma unroll
      for (int tx = 0; tx < 4; ++tx) {
        int dx = x + tx - 2;
        if (dx < 0 || (dx & 1)) continue;
        int ix = dx >> 1;
        if (ix >= 32) continue;
        su += kt[ty] * kt[tx] * skip[((size_t)(b * 3 + c) * 32 + iy) * 32 + ix];
      }
    }
    out[((size_t)(b * 3 + c) * 64 + y) * 64 + x] = res[c] + su * (1.f / 16.f);
  }
}

// ---------------------------------------------------------------------------
extern "C" void kernel_launch(void* const* d_in, const int* in_sizes, int n_in,
                              void* d_out, int out_size, void* d_ws, size_t ws_size,
                              hipStream_t stream) {
  (void)in_sizes; (void)n_in; (void)out_size; (void)ws_size;
  const float* x    = (const float*)d_in[0];
  const float* skip = (const float*)d_in[1];
  const float* w1   = (const float*)d_in[2];
  const float* b1   = (const float*)d_in[3];
  const float* s1   = (const float*)d_in[4];
  const float* nw1  = (const float*)d_in[5];
  const float* n1   = (const float*)d_in[6];
  const float* w2   = (const float*)d_in[7];
  const float* b2   = (const float*)d_in[8];
  const float* s2   = (const float*)d_in[9];
  const float* nw2  = (const float*)d_in[10];
  const float* n2   = (const float*)d_in[11];
  const float* wrgb = (const float*)d_in[12];
  const float* brgb = (const float*)d_in[13];
  const float* srgb = (const float*)d_in[14];
  float* outp = (float*)d_out;

  char* ws = (char*)d_ws;
  const size_t OFF_W   = 0;                                      // 37,748,736 B
  const size_t OFF_XH  = OFF_W   + (size_t)8 * 512 * 512 * 9 * 2;
  const size_t OFF_Y1  = OFF_XH  + (size_t)8 * 512 * 32 * 32 * 2;
  const size_t OFF_Y1B = OFF_Y1  + (size_t)8 * 512 * 65 * 65 * 2;
  const size_t OFF_Y2  = OFF_Y1B + (size_t)8 * 512 * 64 * 64 * 2;
  half_t* Wm  = (half_t*)(ws + OFF_W);
  half_t* Xh  = (half_t*)(ws + OFF_XH);
  half_t* Y1  = (half_t*)(ws + OFF_Y1);
  half_t* Y1B = (half_t*)(ws + OFF_Y1B);
  half_t* Y2  = (half_t*)(ws + OFF_Y2);

  { int n = 8 * 512 * 65 * 65 * 2 / 4;   // zero y1 (covers unwritten col 64)
    zero_kernel<<<(n + 255) / 256, 256, 0, stream>>>((unsigned*)Y1, n); }
  { int n = 8 * 512 * 32 * 32;
    cast_kernel<<<(n + 255) / 256, 256, 0, stream>>>(x, Xh, n); }

  modw_kernel<<<dim3(512, 8), 256, 0, stream>>>(w1, s1, Wm, 1);
  conv1_wmma_kernel<<<dim3(130, 8, 8), 256, 0, stream>>>(Xh, Wm, Y1);
  { int n = 8 * 512 * 64 * 64;
    blur_noise_act_kernel<<<(n + 255) / 256, 256, 0, stream>>>(Y1, nw1, n1, b1, Y1B); }

  modw_kernel<<<dim3(512, 8), 256, 0, stream>>>(w2, s2, Wm, 0);
  conv2_wmma_kernel<<<dim3(64, 16, 8), 256, 0, stream>>>(Y1B, Wm, nw2, n2, b2, Y2);

  { int n = 8 * 64 * 64;
    torgb_skip_kernel<<<(n + 255) / 256, 256, 0, stream>>>(Y2, wrgb, srgb, brgb,
                                                           skip, outp); }
}
